// SelfAttention_45148696215949
// MI455X (gfx1250) — matrix operand order repaired and checked by reading, not compiled, not run on hardware
//
#include <hip/hip_runtime.h>
#include <hip/hip_bf16.h>

#define B_ 8
#define T_ 1024
#define C_ 768
#define H_ 12
#define D_ 64
#define M_ (B_*T_)   // 8192

typedef __attribute__((ext_vector_type(16))) __bf16 bf16x16;
typedef __attribute__((ext_vector_type(8)))  __bf16 bf16x8;
typedef __attribute__((ext_vector_type(8)))  float  v8f;
typedef __attribute__((ext_vector_type(4)))  int    i32x4;

#if defined(__has_builtin)
#if __has_builtin(__builtin_amdgcn_global_load_async_to_lds_b128)
#define HAVE_ASYNC_LDS 1
#endif
#endif
#ifndef HAVE_ASYNC_LDS
#define HAVE_ASYNC_LDS 0
#endif

// b128 async copy operands: i32x4 pointers in AS1 (global) / AS3 (LDS)
typedef __attribute__((address_space(1))) i32x4 as1_i32x4;
typedef __attribute__((address_space(3))) i32x4 as3_i32x4;

static __device__ __forceinline__ as1_i32x4* gptr4(const void* p) {
  return (as1_i32x4*)(uintptr_t)p;
}
static __device__ __forceinline__ as3_i32x4* lptr4(void* p) {
  // generic pointers to LDS carry the byte offset in the low 32 bits
  return (as3_i32x4*)(unsigned)(uintptr_t)p;
}

static __device__ __forceinline__ void wait_async_zero() {
#if HAVE_ASYNC_LDS
#if __has_builtin(__builtin_amdgcn_s_wait_asynccnt)
  __builtin_amdgcn_s_wait_asynccnt(0);
#else
  asm volatile("s_wait_asynccnt 0x0" ::: "memory");
#endif
#endif
}

static __device__ __forceinline__ __bf16 f2bf(float f) {
  unsigned u = __builtin_bit_cast(unsigned, f);
  unsigned r = (u + 0x7FFFu + ((u >> 16) & 1u)) >> 16;   // round-to-nearest-even
  unsigned short s = (unsigned short)r;
  return __builtin_bit_cast(__bf16, s);
}

static __device__ __forceinline__ bf16x16 cat8(bf16x8 a, bf16x8 b) {
  bf16x16 r;
#pragma unroll
  for (int i = 0; i < 8; ++i) { r[i] = a[i]; r[i + 8] = b[i]; }
  return r;
}

static __device__ __forceinline__ v8f vzero() {
  v8f z;
#pragma unroll
  for (int i = 0; i < 8; ++i) z[i] = 0.0f;
  return z;
}

// A fragment: 16 rows x 32 K, row-major source (ld elems).
// Contiguous A k order used here: lane<16 -> M=lane, K=0..15; lane>=16 -> M=lane-16, K=16..31
static __device__ __forceinline__ bf16x16 load_a_frag(const __bf16* base, int ld, int lane) {
  const int half = lane >> 4, lr = lane & 15;
  const __bf16* p = base + (size_t)lr * ld + 16 * half;
  bf16x8 lo = *(const bf16x8*)p;
  bf16x8 hi = *(const bf16x8*)(p + 8);
  return cat8(lo, hi);
}

// B fragment: 32 K x 16 cols; source is Bt row-major [N,K] (row n = column n of B).
// Contiguous B k order used here (matches A): lane<16 -> N=lane, K=0..15; lane>=16 -> N=lane-16, K=16..31
static __device__ __forceinline__ bf16x16 load_b_frag(const __bf16* base, int ld, int lane) {
  const int half = lane >> 4, lr = lane & 15;
  const __bf16* p = base + (size_t)lr * ld + 16 * half;
  bf16x8 lo = *(const bf16x8*)p;
  bf16x8 hi = *(const bf16x8*)(p + 8);
  return cat8(lo, hi);
}

#define WMMA_BF16(a, b, c) \
  __builtin_amdgcn_wmma_f32_16x16x32_bf16(false, (a), false, (b), (short)0, (c), false, false)

static __device__ __forceinline__ float red_max16(float v) {
#pragma unroll
  for (int m = 1; m < 16; m <<= 1) v = fmaxf(v, __shfl_xor(v, m, 32));
  return v;
}
static __device__ __forceinline__ float red_sum16(float v) {
#pragma unroll
  for (int m = 1; m < 16; m <<= 1) v += __shfl_xor(v, m, 32);
  return v;
}

// ---------------- precision conversion kernels ----------------

__global__ __launch_bounds__(256) void cvt_bf16(const float* __restrict__ in,
                                                __bf16* __restrict__ out, int n) {
  int i = blockIdx.x * blockDim.x + threadIdx.x;
  int stride = gridDim.x * blockDim.x;
  for (; i < n; i += stride) out[i] = f2bf(in[i]);
}

// W [K,N] fp32 -> Wt [N,K] bf16
__global__ __launch_bounds__(256) void wtrans_bf16(const float* __restrict__ W,
                                                   __bf16* __restrict__ Wt) {
  int i = blockIdx.x * blockDim.x + threadIdx.x;  // i = n*C_ + k
  if (i >= C_ * C_) return;
  int n = i / C_, k = i - n * C_;
  Wt[i] = f2bf(W[(size_t)k * C_ + n]);
}

// ---------------- WMMA GEMM: out = A[8192,768] @ Bt^T + bias ----------------
// One block: 256 rows x 64 cols; Bt slab (64 x 768) staged once in LDS (async),
// each of 8 waves owns a 32x64 register tile (8 accumulators, 192 WMMAs).
// mode 0: bf16 out, head-major   [B,H,T,D]   (for Q, K)
// mode 1: bf16 out, head-transp. [B,H,D,T]   (for V)
// mode 2: fp32 out, row-major    [M,N]       (final projection)
#define LDS_LD (C_ + 8)   // +16B pad per row: avoids 16-way LDS bank conflicts

__global__ __launch_bounds__(256) void gemm_bf16(const __bf16* __restrict__ A,
                                                 const __bf16* __restrict__ Bt,
                                                 const float* __restrict__ bias,
                                                 void* __restrict__ out,
                                                 int mode) {
  __shared__ __align__(16) __bf16 Bs[64 * LDS_LD];   // ~97 KB (<320 KB/WGP)
  const int tid  = threadIdx.x;
  const int lane = tid & 31;
  const int wv   = tid >> 5;
  const int nb   = blockIdx.x % (C_ / 64);   // N slab (12)
  const int mb   = blockIdx.x / (C_ / 64);   // 256-row slab (32)
  const int K    = C_;

  // ---- stage Bt[nb*64 .. nb*64+64) x [0,768) into LDS ----
  {
    const __bf16* gB = Bt + (size_t)nb * 64 * K;
    for (int c = tid; c < 64 * (K / 8); c += 256) {   // 6144 16-byte chunks
      int row  = c / (K / 8);
      int col8 = c - row * (K / 8);
      const __bf16* src = gB + (size_t)row * K + col8 * 8;
      __bf16* dst = &Bs[row * LDS_LD + col8 * 8];
#if HAVE_ASYNC_LDS
      __builtin_amdgcn_global_load_async_to_lds_b128(gptr4(src), lptr4(dst), 0, 0);
#else
      *(bf16x8*)dst = *(const bf16x8*)src;
#endif
    }
    wait_async_zero();
    __syncthreads();
  }

  const int m0 = mb * 256 + wv * 32;
  const __bf16* arow0 = A + (size_t)m0 * K;
  const __bf16* arow1 = A + (size_t)(m0 + 16) * K;
  const int half = lane >> 4, lr = lane & 15;

  v8f acc[2][4];
#pragma unroll
  for (int s = 0; s < 2; ++s)
#pragma unroll
    for (int j = 0; j < 4; ++j) acc[s][j] = vzero();

  // software pipeline: A fragments for chunk k+1 in flight during chunk k WMMAs
  bf16x16 af0 = load_a_frag(arow0, K, lane);
  bf16x16 af1 = load_a_frag(arow1, K, lane);
  for (int kc = 0; kc < K; kc += 32) {
    const int kn = (kc + 32 < K) ? kc + 32 : 0;
    bf16x16 naf0 = load_a_frag(arow0 + kn, K, lane);
    bf16x16 naf1 = load_a_frag(arow1 + kn, K, lane);
#pragma unroll
    for (int j = 0; j < 4; ++j) {
      const __bf16* p = &Bs[(j * 16 + lr) * LDS_LD + kc + 16 * half];
      bf16x8 lo = *(const bf16x8*)p;
      bf16x8 hi = *(const bf16x8*)(p + 8);
      bf16x16 bfj = cat8(lo, hi);
      acc[0][j] = WMMA_BF16(af0, bfj, acc[0][j]);
      acc[1][j] = WMMA_BF16(af1, bfj, acc[1][j]);
    }
    af0 = naf0;
    af1 = naf1;
  }

#pragma unroll
  for (int j = 0; j < 4; ++j) {
    int n = nb * 64 + j * 16 + lr;
    float bn = bias[n];
#pragma unroll
    for (int s = 0; s < 2; ++s) {
#pragma unroll
      for (int r = 0; r < 8; ++r) {
        int m = m0 + s * 16 + r + 8 * half;   // C layout: VGPR r, half selects +8
        float val = acc[s][j][r] + bn;
        if (mode == 2) {
          ((float*)out)[(size_t)m * C_ + n] = val;
        } else {
          int b = m >> 10, t = m & (T_ - 1);
          int h = n >> 6,  d = n & (D_ - 1);
          size_t idx = (mode == 0)
              ? (((size_t)(b * H_ + h) * T_ + t) * D_ + d)
              : (((size_t)(b * H_ + h) * D_ + d) * T_ + t);
          ((__bf16*)out)[idx] = f2bf(val);
        }
      }
    }
  }
}

// ---------------- flash attention: one wave = 16 query rows of one head ----------------
__global__ __launch_bounds__(256) void attn_fwd(const __bf16* __restrict__ Q,
                                                const __bf16* __restrict__ Kh,
                                                const __bf16* __restrict__ Vt,
                                                __bf16* __restrict__ O) {
  __shared__ __align__(16) __bf16 Pst[8][16 * 32];  // per-wave P staging (C-layout -> A-layout)
  const int lane = threadIdx.x & 31;
  const int wv = threadIdx.x >> 5;
  const int wid = blockIdx.x * 8 + wv;
  const int bh = wid >> 6;                 // 64 q-tiles per head
  const int qt = wid & 63;
  if (bh >= B_ * H_) return;
  const int b = bh / H_, h = bh - b * H_;
  const int q0 = qt * 16;
  const int half = lane >> 4, lr = lane & 15;

  const __bf16* qb = Q  + (size_t)bh * T_ * D_ + (size_t)q0 * D_;
  const __bf16* kb = Kh + (size_t)bh * T_ * D_;
  const __bf16* vb = Vt + (size_t)bh * D_ * T_;

  bf16x16 qf0 = load_a_frag(qb + 0,  D_, lane);   // d = 0..31
  bf16x16 qf1 = load_a_frag(qb + 32, D_, lane);   // d = 32..63

  v8f acc[4] = {vzero(), vzero(), vzero(), vzero()};
  float mi[8], li[8];
#pragma unroll
  for (int r = 0; r < 8; ++r) { mi[r] = -3.0e30f; li[r] = 0.0f; }

  const float sc = 0.125f;                 // 1/sqrt(64)
  for (int s0 = 0; s0 < q0 + 16; s0 += 32) {
    // scores for 32 keys: two 16x16 tiles, K-dim = D = 2 x 32
    bf16x16 k0a = load_b_frag(kb + (size_t)s0 * D_ + 0,  D_, lane);
    bf16x16 k0b = load_b_frag(kb + (size_t)s0 * D_ + 32, D_, lane);
    bf16x16 k1a = load_b_frag(kb + (size_t)(s0 + 16) * D_ + 0,  D_, lane);
    bf16x16 k1b = load_b_frag(kb + (size_t)(s0 + 16) * D_ + 32, D_, lane);
    v8f sA = vzero(); sA = WMMA_BF16(qf0, k0a, sA); sA = WMMA_BF16(qf1, k0b, sA);
    v8f sB = vzero(); sB = WMMA_BF16(qf0, k1a, sB); sB = WMMA_BF16(qf1, k1b, sB);

    const int n0 = s0 + lr;
    const int n1 = s0 + 16 + lr;
    float x0[8], x1[8];
#pragma unroll
    for (int r = 0; r < 8; ++r) {          // causal mask + scale
      int mg = q0 + r + 8 * half;
      x0[r] = (n0 <= mg) ? sA[r] * sc : -3.0e30f;
      x1[r] = (n1 <= mg) ? sB[r] * sc : -3.0e30f;
    }
#pragma unroll
    for (int r = 0; r < 8; ++r) {          // online softmax per row
      float mx = red_max16(fmaxf(x0[r], x1[r]));
      float mnew = fmaxf(mi[r], mx);
      float corr = __expf(mi[r] - mnew);
      float p0 = __expf(x0[r] - mnew);
      float p1 = __expf(x1[r] - mnew);
      float rs = red_sum16(p0 + p1);
      li[r] = li[r] * corr + rs;
      mi[r] = mnew;
#pragma unroll
      for (int j = 0; j < 4; ++j) acc[j][r] *= corr;
      int row = r + 8 * half;
      Pst[wv][row * 32 + lr]      = f2bf(p0);  // P[row][key]
      Pst[wv][row * 32 + lr + 16] = f2bf(p1);
    }
    // re-load P as an A-fragment (lane = M layout) from LDS; DS is in-order per wave
    {
      int rowb = lr * 32 + 16 * half;
      bf16x8 lo = *(const bf16x8*)&Pst[wv][rowb];
      bf16x8 hi = *(const bf16x8*)&Pst[wv][rowb + 8];
      bf16x16 pf = cat8(lo, hi);
#pragma unroll
      for (int j = 0; j < 4; ++j) {        // PV: Vt rows are the B operand
        bf16x16 vf = load_b_frag(vb + (size_t)(j * 16) * T_ + s0, T_, lane);
        acc[j] = WMMA_BF16(pf, vf, acc[j]);
      }
    }
  }
#pragma unroll
  for (int j = 0; j < 4; ++j) {
#pragma unroll
    for (int r = 0; r < 8; ++r) {
      int t = q0 + r + 8 * half;
      int d = h * D_ + j * 16 + lr;
      O[(size_t)(b * T_ + t) * C_ + d] = f2bf(acc[j][r] / li[r]);
    }
  }
}

// ---------------- host launcher ----------------
extern "C" void kernel_launch(void* const* d_in, const int* in_sizes, int n_in,
                              void* d_out, int out_size, void* d_ws, size_t ws_size,
                              hipStream_t stream) {
  (void)in_sizes; (void)n_in; (void)out_size; (void)ws_size;
  const float* x  = (const float*)d_in[0];
  // d_in[1] = mask (recomputed analytically in-kernel)
  const float* Wq = (const float*)d_in[2];
  const float* bq = (const float*)d_in[3];
  const float* Wk = (const float*)d_in[4];
  const float* bk = (const float*)d_in[5];
  const float* Wv = (const float*)d_in[6];
  const float* bv = (const float*)d_in[7];
  const float* Wo = (const float*)d_in[8];
  const float* bo = (const float*)d_in[9];
  float* out = (float*)d_out;

  const size_t SZ_ACT = (size_t)M_ * C_ * sizeof(__bf16);  // 12.58 MB
  const size_t SZ_W   = (size_t)C_ * C_ * sizeof(__bf16);  // 1.18 MB
  char* w = (char*)d_ws;
  __bf16* xb  = (__bf16*)w; w += SZ_ACT;   // x in bf16           [M,C]
  __bf16* qh  = (__bf16*)w; w += SZ_ACT;   // Q head-major        [B,H,T,D]
  __bf16* kh  = (__bf16*)w; w += SZ_ACT;   // K head-major        [B,H,T,D]
  __bf16* vt  = (__bf16*)w; w += SZ_ACT;   // V head-transposed   [B,H,D,T]
  __bf16* ob  = (__bf16*)w; w += SZ_ACT;   // attention output    [M,C]
  __bf16* wqt = (__bf16*)w; w += SZ_W;     // Wq^T bf16 [N,K]
  __bf16* wkt = (__bf16*)w; w += SZ_W;
  __bf16* wvt = (__bf16*)w; w += SZ_W;
  __bf16* wot = (__bf16*)w; w += SZ_W;     // total ~64.5 MB (< 192 MB L2)

  cvt_bf16<<<2048, 256, 0, stream>>>(x, xb, M_ * C_);
  wtrans_bf16<<<(C_ * C_ + 255) / 256, 256, 0, stream>>>(Wq, wqt);
  wtrans_bf16<<<(C_ * C_ + 255) / 256, 256, 0, stream>>>(Wk, wkt);
  wtrans_bf16<<<(C_ * C_ + 255) / 256, 256, 0, stream>>>(Wv, wvt);
  wtrans_bf16<<<(C_ * C_ + 255) / 256, 256, 0, stream>>>(Wo, wot);

  // 384 blocks: 32 M-slabs (256 rows) x 12 N-slabs (64 cols)
  gemm_bf16<<<384, 256, 0, stream>>>(xb, wqt, bq, qh, 0);
  gemm_bf16<<<384, 256, 0, stream>>>(xb, wkt, bk, kh, 0);
  gemm_bf16<<<384, 256, 0, stream>>>(xb, wvt, bv, vt, 1);

  // 6144 waves: 96 heads x 64 query tiles
  attn_fwd<<<768, 256, 0, stream>>>(qh, kh, vt, ob);

  gemm_bf16<<<384, 256, 0, stream>>>(ob, wot, bo, out, 2);
}
